// MixGNN_84378927497242
// MI455X (gfx1250) — matrix operand order repaired and checked by reading, not compiled, not run on hardware
//
#include <hip/hip_runtime.h>
#include <hip/hip_bf16.h>
#include <math.h>

// ---------------------------------------------------------------------------
// MixGNN forward for gfx1250 (wave32, WMMA, LDS-staged GEMM).
//
// Key restructurings vs the reference graph:
//  * GCN linearity: Â(X W) == (Â X) W  -> scatter-add runs in the 64-dim
//    input space (halves edge gather/atomic traffic, the dominant cost),
//    then a single WMMA GEMM lifts to 128 dims.
//  * All pre-BatchNorm biases dropped (BN mean-subtraction cancels them).
//  * GEMMs: v_wmma_f32_16x16x32_f16, A/B chunks staged in LDS (ds_load_b128
//    fragment reads), f32 accumulation.
// ---------------------------------------------------------------------------

typedef _Float16 v16h __attribute__((ext_vector_type(16)));
typedef float    v8f  __attribute__((ext_vector_type(8)));

#define GNN_INP 64
#define GNN_HID 128
#define GEMM_WAVES 8

// ---------------------------------------------------------------------------
// C[M,N] = A[M,K] @ B[K,N] via WMMA f16 fragments, f32 accum.
// Requires M%16==0, N%16==0 with N<=256, K%32==0. Block = 256 thr (8 waves)
// owns one 16-row stripe; waves split the <=16 column tiles (<=2 per wave).
// Per K-chunk: stage A(16x32) and B^T(Nx32) in LDS, then WMMA from LDS.
//
// Fragment layouts per CDNA5 ISA 7.12.2 (wave32):
//  A 16x32 f16 : lane(0..15)=row M; halves j<8 -> K=lhalf*8+j,
//                j>=8 -> K=lhalf*8+16+(j-8)  (two 16B runs per lane).
//  B 32x16 f16 : lane(0..15)=col N; halves j<8 -> K=lhalf*8+j,
//                j>=8 -> K=lhalf*8+16+(j-8) (two 16B runs per lane, transposed LDS image).
//  C/D 16x16 f32: lane -> col, VGPR v -> row v + 8*lanehalf.
// ---------------------------------------------------------------------------
__global__ __launch_bounds__(256) void k_wmma_gemm(
    const float* __restrict__ A, const float* __restrict__ Bm,
    float* __restrict__ C, int M, int K, int N)
{
    __shared__ _Float16 As[16 * 32];    // [row][k]  1 KB
    __shared__ _Float16 Bs[256 * 32];   // [col][k] 16 KB (N <= 256)

    const int lane  = threadIdx.x & 31;
    const int wave  = threadIdx.x >> 5;
    const int lrow  = lane & 15;
    const int lhalf = lane >> 4;
    const int NT    = N >> 4;
    const int r0    = blockIdx.x << 4;
    if (r0 >= M) return;

    v8f acc0 = {}, acc1 = {};

    for (int kk = 0; kk < K; kk += 32) {
        // ---- stage A chunk [16][32] f32->f16, coalesced over k ----
        for (int idx = threadIdx.x; idx < 16 * 32; idx += 256) {
            int r = idx >> 5, k = idx & 31;
            As[idx] = (_Float16)A[(size_t)(r0 + r) * K + kk + k];
        }
        // ---- stage B chunk transposed: Bs[c*32+k] = B[(kk+k)*N+c] ----
        for (int idx = threadIdx.x; idx < N * 32; idx += 256) {
            int k = idx / N, c = idx - k * N;       // consecutive idx -> consecutive c
            Bs[c * 32 + k] = (_Float16)Bm[(size_t)(kk + k) * N + c];
        }
        __syncthreads();

        v16h afrag;
        const _Float16* ap = As + lrow * 32 + lhalf * 8;
#pragma unroll
        for (int j = 0; j < 8; ++j) { afrag[j] = ap[j]; afrag[8 + j] = ap[16 + j]; }

#pragma unroll
        for (int i = 0; i < 2; ++i) {
            int cn = wave + i * GEMM_WAVES;
            if (cn < NT) {
                v16h bfrag;
                const _Float16* bp = Bs + ((cn << 4) + lrow) * 32 + lhalf * 8;
#pragma unroll
                for (int j = 0; j < 8; ++j) { bfrag[j] = bp[j]; bfrag[8 + j] = bp[16 + j]; }
#if defined(__HIP_DEVICE_COMPILE__)
                v8f r = __builtin_amdgcn_wmma_f32_16x16x32_f16(
                    false, afrag, false, bfrag, (short)0,
                    (i == 0) ? acc0 : acc1, false, false);
                if (i == 0) acc0 = r; else acc1 = r;
#endif
            }
        }
        __syncthreads();
    }

#pragma unroll
    for (int i = 0; i < 2; ++i) {
        int cn = wave + i * GEMM_WAVES;
        if (cn < NT) {
            v8f acc = (i == 0) ? acc0 : acc1;
            int col = (cn << 4) + lrow;
#pragma unroll
            for (int v = 0; v < 8; ++v)
                C[(size_t)(r0 + v + lhalf * 8) * N + col] = acc[v];
        }
    }
}

// ------------------------- degree / normalization --------------------------
__global__ void k_deg_init(float* __restrict__ deg, int N) {
    int i = blockIdx.x * blockDim.x + threadIdx.x;
    if (i < N) deg[i] = 1.0f;                       // self loop
}
__global__ void k_deg_edges(const int* __restrict__ dst, float* __restrict__ deg, int E) {
    int e = blockIdx.x * blockDim.x + threadIdx.x;
    if (e < E) atomicAdd(&deg[dst[e]], 1.0f);
}
__global__ void k_rsqrt(float* __restrict__ d, int N) {
    int i = blockIdx.x * blockDim.x + threadIdx.x;
    if (i < N) d[i] = rsqrtf(d[i]);
}

// aggx = self-loop term in INPUT space: x[i,:] * dis[i]^2   (64 features)
__global__ void k_aggx_self(const float* __restrict__ x, const float* __restrict__ dis,
                            float* __restrict__ aggx, int N) {
    long gid = (long)blockIdx.x * blockDim.x + threadIdx.x;
    if (gid >= (long)N * GNN_INP) return;
    int i = (int)(gid >> 6);
    float w = dis[i] * dis[i];
    aggx[gid] = x[gid] * w;
}

// 16 threads per edge, 4 features each (64 feats): float4 gather + f32 atomics.
__global__ void k_scatter_edges(const int* __restrict__ src, const int* __restrict__ dst,
                                const float* __restrict__ dis, const float* __restrict__ x,
                                float* __restrict__ aggx, int E) {
    long gid = (long)blockIdx.x * blockDim.x + threadIdx.x;
    int e = (int)(gid >> 4);
    if (e >= E) return;
    int f = (int)(gid & 15) * 4;
    int s = src[e], d = dst[e];
    float w = dis[s] * dis[d];
    const float4 hv = *(const float4*)(x + (size_t)s * GNN_INP + f);
    float* ap = aggx + (size_t)d * GNN_INP + f;
    atomicAdd(ap + 0, hv.x * w);
    atomicAdd(ap + 1, hv.y * w);
    atomicAdd(ap + 2, hv.z * w);
    atomicAdd(ap + 3, hv.w * w);
}

// ------------------------------ BatchNorm ----------------------------------
__global__ void k_zero(float* __restrict__ p, int n) {
    int i = blockIdx.x * blockDim.x + threadIdx.x;
    if (i < n) p[i] = 0.0f;
}

// blockDim.x == F; thread t owns feature t; rows strided by gridDim.x (coalesced).
__global__ void k_bn_stats(const float* __restrict__ X, float* __restrict__ sums,
                           float* __restrict__ sqs, int R, int F) {
    int f = threadIdx.x;
    if (f >= F) return;
    float s = 0.f, q = 0.f;
    for (int r = blockIdx.x; r < R; r += gridDim.x) {
        float v = X[(size_t)r * F + f];
        s += v; q += v * v;
    }
    atomicAdd(&sums[f], s);
    atomicAdd(&sqs[f], q);
}

__global__ void k_bn_finalize(const float* __restrict__ sums, const float* __restrict__ sqs,
                              const float* __restrict__ gamma, const float* __restrict__ beta,
                              float* __restrict__ scale, float* __restrict__ shift,
                              int R, int F) {
    int f = threadIdx.x + blockIdx.x * blockDim.x;
    if (f >= F) return;
    float m  = sums[f] / (float)R;
    float v  = sqs[f] / (float)R - m * m;
    float sc = gamma[f] * rsqrtf(v + 1e-5f);
    scale[f] = sc;
    shift[f] = beta[f] - m * sc;
}

// mode: 0=bn, 1=bn+relu, 2=bn+relu then ^2, 3=bn+relu then ^4
__global__ void k_bn_apply(const float* __restrict__ X, const float* __restrict__ scale,
                           const float* __restrict__ shift, float* __restrict__ Y,
                           int R, int F, int mode) {
    long gid = (long)blockIdx.x * blockDim.x + threadIdx.x;
    if (gid >= (long)R * F) return;
    int f = (int)(gid % F);
    float t = X[gid] * scale[f] + shift[f];
    if (mode >= 1) t = fmaxf(t, 0.0f);
    if (mode == 2) t = t * t;
    if (mode == 3) { t = t * t; t = t * t; }
    Y[gid] = t;
}

// BN1 + relu fused with global mean-pool scatter (F = GNN_HID = 128).
__global__ void k_bn_relu_pool(const float* __restrict__ agg, const float* __restrict__ scale,
                               const float* __restrict__ shift, const int* __restrict__ batch,
                               float* __restrict__ pool, float* __restrict__ cnt, int N) {
    long gid = (long)blockIdx.x * blockDim.x + threadIdx.x;
    if (gid >= (long)N * GNN_HID) return;
    int i = (int)(gid >> 7), f = (int)(gid & 127);
    float t = fmaxf(agg[gid] * scale[f] + shift[f], 0.0f);
    int g = batch[i];
    atomicAdd(&pool[(size_t)g * GNN_HID + f], t);
    if (f == 0) atomicAdd(&cnt[g], 1.0f);
}

__global__ void k_pool_div(float* __restrict__ pool, const float* __restrict__ cnt, int G) {
    long gid = (long)blockIdx.x * blockDim.x + threadIdx.x;
    if (gid >= (long)G * GNN_HID) return;
    pool[gid] /= fmaxf(cnt[gid >> 7], 1.0f);
}

// t[b, 0:64] = x[b]; t[b, 64:128] = z2[masks[b]]
__global__ void k_concat(const float* __restrict__ x, const float* __restrict__ z2,
                         const int* __restrict__ masks, float* __restrict__ t, int B) {
    long gid = (long)blockIdx.x * blockDim.x + threadIdx.x;
    if (gid >= (long)B * 128) return;
    int b = (int)(gid >> 7), f = (int)(gid & 127);
    t[gid] = (f < 64) ? x[(size_t)b * 64 + f]
                      : z2[(size_t)masks[b] * 64 + (f - 64)];
}

// out[b] = dot(a2[b,:64], W_fc3[:,0]) + b_fc3
__global__ void k_fc3(const float* __restrict__ a2, const float* __restrict__ W,
                      const float* __restrict__ bias, float* __restrict__ out, int B) {
    int b = blockIdx.x * blockDim.x + threadIdx.x;
    if (b >= B) return;
    float s = 0.f;
#pragma unroll 8
    for (int k = 0; k < 64; ++k) s += a2[(size_t)b * 64 + k] * W[k];
    out[b] = s + bias[0];
}

// ---------------------------------------------------------------------------
extern "C" void kernel_launch(void* const* d_in, const int* in_sizes, int n_in,
                              void* d_out, int out_size, void* d_ws, size_t ws_size,
                              hipStream_t stream) {
    (void)n_in; (void)out_size; (void)ws_size;

    const float* x      = (const float*)d_in[0];
    const int*   masks  = (const int*)  d_in[1];
    const float* gnn_x  = (const float*)d_in[2];
    const int*   eidx   = (const int*)  d_in[3];
    const int*   gbatch = (const int*)  d_in[4];
    const float* W_gcn  = (const float*)d_in[5];
    const float* g1  = (const float*)d_in[7],  *be1 = (const float*)d_in[8];
    const float* W_lin1 = (const float*)d_in[9];
    const float* g2  = (const float*)d_in[11], *be2 = (const float*)d_in[12];
    const float* W_lin2 = (const float*)d_in[13];
    const float* g3  = (const float*)d_in[15], *be3 = (const float*)d_in[16];
    const float* W_fc1  = (const float*)d_in[17];
    const float* g4  = (const float*)d_in[19], *be4 = (const float*)d_in[20];
    const float* W_fc2  = (const float*)d_in[21];
    const float* g5  = (const float*)d_in[23], *be5 = (const float*)d_in[24];
    const float* W_fc3  = (const float*)d_in[25];
    const float* b_fc3  = (const float*)d_in[26];
    float* out = (float*)d_out;

    const int N   = in_sizes[4];       // 100000 nodes
    const int E   = in_sizes[3] / 2;   // 1.6M edges
    const int Bsz = in_sizes[1];       // 1024
    const int G   = 1024;              // N_GRAPHS
    const int* e_src = eidx;
    const int* e_dst = eidx + E;

    // ---- workspace carve-out (all f32) ----
    float* ws = (float*)d_ws; size_t off = 0;
    float* aggx = ws + off; off += (size_t)N * GNN_INP;   // (ÂX)  [N,64]
    float* dis  = ws + off; off += (size_t)N;
    float* agg  = ws + off; off += (size_t)N * GNN_HID;   // (ÂX)W [N,128]
    float* pool = ws + off; off += (size_t)G * GNN_HID;   // pool then cnt: contiguous
    float* cnt  = ws + off; off += (size_t)G;
    float* y1   = ws + off; off += (size_t)G * 64;
    float* z1   = ws + off; off += (size_t)G * 64;
    float* y2   = ws + off; off += (size_t)G * 64;
    float* z2   = ws + off; off += (size_t)G * 64;
    float* tc   = ws + off; off += (size_t)Bsz * 128;
    float* y3   = ws + off; off += (size_t)Bsz * 256;
    float* a1   = ws + off; off += (size_t)Bsz * 256;
    float* y4   = ws + off; off += (size_t)Bsz * 64;
    float* a2   = ws + off; off += (size_t)Bsz * 64;
    float* sums = ws + off; off += 256;                   // sums+sqs contiguous
    float* sqs  = ws + off; off += 256;
    float* scl  = ws + off; off += 256;
    float* shf  = ws + off; off += 256;

    auto cdiv = [](long a, long b) { return (int)((a + b - 1) / b); };

    // 1) symmetric normalization: deg (incl self loop) -> rsqrt
    k_deg_init <<<cdiv(N, 256), 256, 0, stream>>>(dis, N);
    k_deg_edges<<<cdiv(E, 256), 256, 0, stream>>>(e_dst, dis, E);
    k_rsqrt    <<<cdiv(N, 256), 256, 0, stream>>>(dis, N);

    // 2) aggx = Â X  (scatter in 64-dim input space: half the edge traffic)
    k_aggx_self    <<<cdiv((long)N * GNN_INP, 256), 256, 0, stream>>>(gnn_x, dis, aggx, N);
    k_scatter_edges<<<cdiv((long)E * 16, 256), 256, 0, stream>>>(e_src, e_dst, dis, gnn_x, aggx, E);

    // 3) agg = aggx @ W_gcn  [N,64]x[64,128]  (b_gcn absorbed by BN1)
    k_wmma_gemm<<<N / 16, 256, 0, stream>>>(aggx, W_gcn, agg, N, 64, 128);

    // 4) BN1 + relu fused with mean-pool
    k_zero       <<<2, 256, 0, stream>>>(sums, 512);
    k_bn_stats   <<<800, 128, 0, stream>>>(agg, sums, sqs, N, 128);
    k_bn_finalize<<<1, 128, 0, stream>>>(sums, sqs, g1, be1, scl, shf, N, 128);
    k_zero       <<<cdiv(G * GNN_HID + G, 256), 256, 0, stream>>>(pool, G * GNN_HID + G);
    k_bn_relu_pool<<<cdiv((long)N * GNN_HID, 256), 256, 0, stream>>>(
        agg, scl, shf, gbatch, pool, cnt, N);
    k_pool_div   <<<cdiv(G * GNN_HID, 256), 256, 0, stream>>>(pool, cnt, G);

    // 5) y1 = pool @ W_lin1 [1024,128]x[128,64]; BN2 + relu -> z1
    k_wmma_gemm  <<<G / 16, 256, 0, stream>>>(pool, W_lin1, y1, G, 128, 64);
    k_zero       <<<2, 256, 0, stream>>>(sums, 512);
    k_bn_stats   <<<256, 64, 0, stream>>>(y1, sums, sqs, G, 64);
    k_bn_finalize<<<1, 64, 0, stream>>>(sums, sqs, g2, be2, scl, shf, G, 64);
    k_bn_apply   <<<cdiv(G * 64, 256), 256, 0, stream>>>(y1, scl, shf, z1, G, 64, 1);

    // 6) y2 = z1 @ W_lin2 [1024,64]x[64,64]; BN3 (no relu) -> z2
    k_wmma_gemm  <<<G / 16, 256, 0, stream>>>(z1, W_lin2, y2, G, 64, 64);
    k_zero       <<<2, 256, 0, stream>>>(sums, 512);
    k_bn_stats   <<<256, 64, 0, stream>>>(y2, sums, sqs, G, 64);
    k_bn_finalize<<<1, 64, 0, stream>>>(sums, sqs, g3, be3, scl, shf, G, 64);
    k_bn_apply   <<<cdiv(G * 64, 256), 256, 0, stream>>>(y2, scl, shf, z2, G, 64, 0);

    // 7) gather + concat: tc = [x | z2[masks]]
    k_concat<<<cdiv((long)Bsz * 128, 256), 256, 0, stream>>>(x, z2, masks, tc, Bsz);

    // 8) y3 = tc @ W_fc1 [1024,128]x[128,256]; BN4 + relu + ^2 -> a1
    k_wmma_gemm  <<<Bsz / 16, 256, 0, stream>>>(tc, W_fc1, y3, Bsz, 128, 256);
    k_zero       <<<2, 256, 0, stream>>>(sums, 512);
    k_bn_stats   <<<256, 256, 0, stream>>>(y3, sums, sqs, Bsz, 256);
    k_bn_finalize<<<1, 256, 0, stream>>>(sums, sqs, g4, be4, scl, shf, Bsz, 256);
    k_bn_apply   <<<cdiv((long)Bsz * 256, 256), 256, 0, stream>>>(y3, scl, shf, a1, Bsz, 256, 2);

    // 9) y4 = a1 @ W_fc2 [1024,256]x[256,64]; BN5 + relu + ^4 -> a2
    k_wmma_gemm  <<<Bsz / 16, 256, 0, stream>>>(a1, W_fc2, y4, Bsz, 256, 64);
    k_zero       <<<2, 256, 0, stream>>>(sums, 512);
    k_bn_stats   <<<256, 64, 0, stream>>>(y4, sums, sqs, Bsz, 64);
    k_bn_finalize<<<1, 64, 0, stream>>>(sums, sqs, g5, be5, scl, shf, Bsz, 64);
    k_bn_apply   <<<cdiv(Bsz * 64, 256), 256, 0, stream>>>(y4, scl, shf, a2, Bsz, 64, 3);

    // 10) out = a2 @ W_fc3 + b_fc3   [1024,64]x[64,1]
    k_fc3<<<cdiv(Bsz, 256), 256, 0, stream>>>(a2, W_fc3, b_fc3, out, Bsz);
}
